// Cortex_35588099014796
// MI455X (gfx1250) — compile-verified, matrix operand order checked by reading, not run on hardware
//
#include <hip/hip_runtime.h>
#include <hip/hip_bf16.h>
#include <math.h>

typedef __attribute__((ext_vector_type(16))) _Float16 v16h;
typedef __attribute__((ext_vector_type(8)))  float    v8f;

#define TT   512
#define INN  1024
#define ISD  32
#define DD   256
#define NOUT 128
#define KK   (2 * DD * DD)        // 131072
#define KSPLIT 8
#define NWG  16                   // workgroups in the scan cohort (== max cluster size)

// ---------------------------------------------------------------------------
// 0) init: zero S plane 0 and the barrier counter (ws is poisoned, not zeroed)
// ---------------------------------------------------------------------------
__global__ void init_kernel(float* __restrict__ S0, unsigned* __restrict__ cnt) {
  int i = blockIdx.x * blockDim.x + threadIdx.x;
  int stride = gridDim.x * blockDim.x;
  for (; i < DD * DD; i += stride) S0[i] = 0.0f;
  if (blockIdx.x == 0 && threadIdx.x == 0) cnt[0] = 0u;
}

// ---------------------------------------------------------------------------
// 1) embedding: emb[t][i] = sum_j X[t][j] * We[i][j]   (exact f32, LDS-tiled)
// ---------------------------------------------------------------------------
__global__ __launch_bounds__(256) void embed_kernel(const float* __restrict__ X,
                                                    const float* __restrict__ We,
                                                    float* __restrict__ emb) {
  __shared__ float Xs[16][17];
  __shared__ float Ws[16][17];
  const int tx = threadIdx.x & 15;
  const int ty = threadIdx.x >> 4;
  const int i0 = blockIdx.x * 16;   // 64 blocks in x  -> 1024 cols
  const int t0 = blockIdx.y * 16;   // 32 blocks in y  -> 512 rows
  float acc = 0.0f;
  for (int k0 = 0; k0 < INN; k0 += 16) {
    Xs[ty][tx] = X[(t0 + ty) * INN + k0 + tx];
    Ws[ty][tx] = We[(i0 + ty) * INN + k0 + tx];
    __syncthreads();
#pragma unroll
    for (int kk = 0; kk < 16; ++kk) acc += Xs[ty][kk] * Ws[tx][kk];
    __syncthreads();
  }
  emb[(t0 + ty) * INN + i0 + tx] = acc;
}

// ---------------------------------------------------------------------------
// 2) W_out f32 -> f16 (keeps GEMM working set inside the 192MB L2)
// ---------------------------------------------------------------------------
__global__ void cvt_f16_kernel(const float* __restrict__ src,
                               _Float16* __restrict__ dst, int n) {
  int i = blockIdx.x * blockDim.x + threadIdx.x;
  int stride = gridDim.x * blockDim.x;
  for (; i < n; i += stride) dst[i] = (_Float16)src[i];
}

// ---------------------------------------------------------------------------
// 3) reservoir scan: 16-WG cohort, per-step device barrier.
//    V state in registers, S double-buffered in global (L2-resident),
//    separable circular rowsums staged in LDS.
// ---------------------------------------------------------------------------
__global__ __launch_bounds__(1024) void scan_kernel(const float* __restrict__ emb,
                                                    const float* __restrict__ maskc,
                                                    const float* __restrict__ maskf,
                                                    float* __restrict__ Sbuf,
                                                    _Float16* __restrict__ outs,
                                                    unsigned* __restrict__ cnt) {
  __shared__ float rs5[20][DD];   // rowsum of 5-tap  for rows r0-2 .. r0+17
  __shared__ float rs9[32][DD];   // rowsum of 9-tap(d2) for rows r0-8 .. r0+23
  __shared__ float uc[2][ISD];    // tanh'd coarse input, 2 coarse rows

  const int wg  = blockIdx.x;       // 0..15, owns 16 rows
  const int tid = threadIdx.x;      // 0..1023
  const int r0  = wg * 16;
  const int x   = tid & 255;
  const int yb  = tid >> 8;         // 0..3
  const int cy0 = wg * 2;

  float V[4];
  float mf[4];
#pragma unroll
  for (int i = 0; i < 4; ++i) {
    V[i] = 0.0f;
    int gy = r0 + yb + 4 * i;
    mf[i] = maskf[gy * DD + x];
  }

  for (int t = 0; t < TT; ++t) {
    const float* __restrict__ Sold = Sbuf + (t & 1) * (DD * DD);
    float* __restrict__ Snew       = Sbuf + ((t + 1) & 1) * (DD * DD);

    // coarse drive for this WG's 2 coarse rows
    if (tid < 64) {
      int j = tid >> 5, cx = tid & 31;
      float e = emb[t * INN + (cy0 + j) * ISD + cx] * maskc[(cy0 + j) * ISD + cx];
      uc[j][cx] = tanhf(e);   // SCALE == 1
    }
    // rowsum, 9 taps dilation 2, 32 rows (8192 entries / 1024 threads)
#pragma unroll
    for (int i = 0; i < 8; ++i) {
      int e = tid + i * 1024;
      int rl = e >> 8, xx = e & 255;
      const float* row = Sold + ((r0 - 8 + rl) & 255) * DD;
      float s = 0.0f;
#pragma unroll
      for (int d = -8; d <= 8; d += 2) s += row[(xx + d) & 255];
      rs9[rl][xx] = s;
    }
    // rowsum, 5 taps, 20 rows (5120 entries / 1024 threads)
#pragma unroll
    for (int i = 0; i < 5; ++i) {
      int e = tid + i * 1024;
      int rl = e >> 8, xx = e & 255;
      const float* row = Sold + ((r0 - 2 + rl) & 255) * DD;
      float s = 0.0f;
#pragma unroll
      for (int d = -2; d <= 2; ++d) s += row[(xx + d) & 255];
      rs5[rl][xx] = s;
    }
    __syncthreads();

    // neuron update for this thread's 4 pixels
#pragma unroll
    for (int i = 0; i < 4; ++i) {
      int yl = yb + 4 * i;
      int gy = r0 + yl;
      float s5 = 0.0f, s9 = 0.0f;
#pragma unroll
      for (int d = 0; d < 5; ++d) s5 += rs5[yl + d][x];
#pragma unroll
      for (int d = 0; d < 9; ++d) s9 += rs9[yl + 2 * d][x];
      float lat = (1.0f / 25.0f) * s5 + (-0.5f / 81.0f) * s9;  // EXC*avg5 + INH*avg9
      float xt  = uc[yl >> 3][x >> 3] * mf[i];
      float v   = 0.9f * V[i] + 0.5f * xt;                     // DECAY, SPLIT
      if (v >= 0.1f) v += 0.5f * xt + lat;                     // LOWER
      v = fminf(v, 1.0f);
      float sb = (v > 0.75f) ? 1.0f : 0.0f;                    // FIRE
      if (v > 0.75f) v = 0.0f;                                 // RESET
      V[i] = v;
      Snew[gy * DD + x] = sb;
      size_t base = (size_t)t * KK + (size_t)gy * DD + x;      // outs[t][0]=V
      outs[base]           = (_Float16)v;
      outs[base + DD * DD] = (_Float16)sb;                     // outs[t][1]=S
    }

    // device-wide step barrier: release -> arrive -> acquire
    __threadfence();
    __syncthreads();
    if (tid == 0) {
      unsigned target = (unsigned)(t + 1) * gridDim.x;
      atomicAdd(cnt, 1u);
      while (__hip_atomic_load(cnt, __ATOMIC_ACQUIRE, __HIP_MEMORY_SCOPE_AGENT) < target) {}
    }
    __builtin_amdgcn_s_cluster_barrier();  // s_barrier_signal/-wait -3 (NOP if unclustered)
    __syncthreads();
    __threadfence();
  }
}

// ---------------------------------------------------------------------------
// 4) readout GEMM via WMMA f16->f32: partial[kc][m][n] over K/8 chunks.
//    One wave per 16x16 output tile per K-chunk. B column n == W_out row n
//    (contiguous); A/B operands use the same K-permutation so lane loads are
//    plain 32B contiguous reads.
// ---------------------------------------------------------------------------
__global__ __launch_bounds__(256) void readout_gemm(const _Float16* __restrict__ A,
                                                    const _Float16* __restrict__ B,
                                                    float* __restrict__ partial) {
  const int wave = threadIdx.x >> 5;     // 8 waves -> N tiles (N=128)
  const int lane = threadIdx.x & 31;
  const int m0   = blockIdx.x * 16;      // 32 blocks -> M=512
  const int n0   = wave * 16;
  const int kc   = blockIdx.y;           // 8 K-chunks
  const int kbeg = kc * (KK / KSPLIT);
  const int kend = kbeg + (KK / KSPLIT);
  const int half = lane >> 4;            // 0: k..k+15, 1: k+16..k+31
  const int r    = lane & 15;

  const _Float16* Ap = A + (size_t)(m0 + r) * KK + half * 16;
  const _Float16* Bp = B + (size_t)(n0 + r) * KK + half * 16;

  v8f acc = {};
  for (int k = kbeg; k < kend; k += 128) {
    __builtin_prefetch((const void*)(Ap + k + 1024), 0, 3);
    __builtin_prefetch((const void*)(Bp + k + 1024), 0, 3);
#pragma unroll
    for (int u = 0; u < 4; ++u) {
      int kk = k + u * 32;
      v16h a = *(const v16h*)(Ap + kk);
      v16h b = *(const v16h*)(Bp + kk);
      acc = __builtin_amdgcn_wmma_f32_16x16x32_f16(
          /*neg_a=*/false, a, /*neg_b=*/false, b,
          /*c_mod=*/(short)0, acc, /*reuse_a=*/false, /*reuse_b=*/false);
    }
  }
  // C/D layout: VGPR v -> M = m0 + v + 8*half, N = n0 + (lane&15)
  float* out = partial + ((size_t)kc * TT + (m0 + half * 8)) * NOUT + (n0 + r);
#pragma unroll
  for (int v = 0; v < 8; ++v) out[(size_t)v * NOUT] = acc[v];
}

// ---------------------------------------------------------------------------
// 5) reduce K-chunks + bias
// ---------------------------------------------------------------------------
__global__ void reduce_kernel(const float* __restrict__ partial,
                              const float* __restrict__ b,
                              float* __restrict__ Y) {
  int i = blockIdx.x * blockDim.x + threadIdx.x;   // 512*128
  if (i < TT * NOUT) {
    float s = b[i & (NOUT - 1)];
#pragma unroll
    for (int r2 = 0; r2 < KSPLIT; ++r2) s += partial[(size_t)r2 * TT * NOUT + i];
    Y[i] = s;
  }
}

// ---------------------------------------------------------------------------
extern "C" void kernel_launch(void* const* d_in, const int* in_sizes, int n_in,
                              void* d_out, int out_size, void* d_ws, size_t ws_size,
                              hipStream_t stream) {
  (void)in_sizes; (void)n_in; (void)out_size; (void)ws_size;
  const float* X     = (const float*)d_in[0];  // [512,1024]
  const float* We    = (const float*)d_in[1];  // [1024,1024]
  const float* maskc = (const float*)d_in[2];  // [32,32]
  const float* maskf = (const float*)d_in[3];  // [256,256]
  const float* W_out = (const float*)d_in[4];  // [128, 2*256*256]
  const float* b_out = (const float*)d_in[5];  // [128]
  float* Y = (float*)d_out;                    // [512,128]

  size_t off = 0;
  auto carve = [&](size_t bytes) -> void* {
    void* p = (char*)d_ws + off;
    off += (bytes + 255) & ~(size_t)255;
    return p;
  };
  _Float16* outs_h  = (_Float16*)carve((size_t)TT * KK * sizeof(_Float16));      // 128 MiB
  _Float16* Wh      = (_Float16*)carve((size_t)NOUT * KK * sizeof(_Float16));    //  32 MiB
  float*    emb     = (float*)carve((size_t)TT * INN * sizeof(float));           //   2 MiB
  float*    partial = (float*)carve((size_t)KSPLIT * TT * NOUT * sizeof(float)); //   2 MiB
  float*    Sbuf    = (float*)carve((size_t)2 * DD * DD * sizeof(float));        // 512 KiB
  unsigned* cnt     = (unsigned*)carve(256);

  init_kernel<<<64, 256, 0, stream>>>(Sbuf, cnt);
  embed_kernel<<<dim3(INN / 16, TT / 16), 256, 0, stream>>>(X, We, emb);
  cvt_f16_kernel<<<4096, 256, 0, stream>>>(W_out, Wh, NOUT * KK);
  scan_kernel<<<NWG, 1024, 0, stream>>>(emb, maskc, maskf, Sbuf, outs_h, cnt);
  readout_gemm<<<dim3(TT / 16, KSPLIT), 256, 0, stream>>>(outs_h, Wh, partial);
  reduce_kernel<<<(TT * NOUT + 255) / 256, 256, 0, stream>>>(partial, b_out, Y);
}
